// Block_19095424598462
// MI455X (gfx1250) — matrix operand order repaired and checked by reading, not compiled, not run on hardware
//
#include <hip/hip_runtime.h>
#include <math.h>
#include <stdint.h>

// ---------------------------------------------------------------------------
// MI455X (gfx1250) talking-heads transformer block.
// All matmuls: v_wmma_f32_16x16x32_bf16 (wave32 WMMA), f32 accumulation.
// Activations bf16 in workspace; q/k/vT/ctx (~19MB each) are L2-resident
// (192MB L2). Q tiles are staged in LDS and read per-iteration as
// ds_load_b128; an asm memory clobber in the streaming loops stops LICM from
// hoisting the 18-fragment (144 VGPR) loop-invariant set into scratch spills
// (round-1 failure) without the SCOPE_SYS flat-load serialization that
// `volatile` caused (round-2 failure).
// ---------------------------------------------------------------------------

typedef __attribute__((ext_vector_type(16))) __bf16 v16bf;
typedef __attribute__((ext_vector_type(8)))  float  v8f;
typedef __attribute__((ext_vector_type(4)))  unsigned int v4u;

union FragU { v4u q[2]; v16bf v; };
union Pack8 { v4u q; unsigned short h[8]; };

// Compiler-level barrier: pretends to write memory, so loop-invariant loads
// cannot be hoisted across it; costs zero instructions.
#define NO_LICM() asm volatile("" ::: "memory")

__device__ __forceinline__ unsigned short f2bf(float f) {
  unsigned int u = __float_as_uint(f);
  u += 0x7fffu + ((u >> 16) & 1u);          // round-to-nearest-even
  return (unsigned short)(u >> 16);
}

__device__ __forceinline__ v8f wmma_bf16(v16bf a, v16bf b, v8f c) {
  return __builtin_amdgcn_wmma_f32_16x16x32_bf16(false, a, false, b,
                                                 (short)0, c, false, false);
}

// A fragment: 16x32 bf16 tile from row-major global memory (ld elems/row).
// Lane layout (ISA 7.12.2): m=lane%16, half=lane/16; elems 0..7 = K(half*8..+7),
// elems 8..15 = K(16+half*8..+7).
__device__ __forceinline__ v16bf load_a_frag(const unsigned short* base, int ld,
                                             int row0, int col0) {
  const int lane = threadIdx.x & 31;
  const int m = lane & 15, hf = lane >> 4;
  const unsigned short* p = base + (size_t)(row0 + m) * ld + col0 + hf * 8;
  FragU f;
  f.q[0] = *(const v4u*)p;
  f.q[1] = *(const v4u*)(p + 16);
  return f.v;
}

// Same A-fragment pattern but from an LDS tile (plain loads -> ds_load_b128).
__device__ __forceinline__ v16bf load_a_frag_sh(const unsigned short* base,
                                                int ld, int col0) {
  const int lane = threadIdx.x & 31;
  const int m = lane & 15, hf = lane >> 4;
  const unsigned short* p = base + m * ld + col0 + hf * 8;
  FragU f;
  f.q[0] = *(const v4u*)p;
  f.q[1] = *(const v4u*)(p + 16);
  return f.v;
}

// B fragment: 32x16 bf16 (K x N), source holds B^T row-major [N, K].
// Lane layout: n=lane%16, half=lane/16; elems 0..7 = K(half*8..+7),
// elems 8..15 = K(16+half*8..+7).
__device__ __forceinline__ v16bf load_b_frag(const unsigned short* baseT, int ld,
                                             int ncol0, int k0) {
  const int lane = threadIdx.x & 31;
  const int n = lane & 15, hf = lane >> 4;
  const unsigned short* p = baseT + (size_t)(ncol0 + n) * ld + k0 + hf * 8;
  FragU f;
  f.q[0] = *(const v4u*)p;
  f.q[1] = *(const v4u*)(p + 16);
  return f.v;
}

#define VZERO8 {0.f,0.f,0.f,0.f,0.f,0.f,0.f,0.f}
#define QLD 584   // 576 + 8 pad: 16-lane A-frag reads are bank-conflict-free
#define PMLD 40   // 32 + 8 pad

// ---------------------------------------------------------------------------
// Weight transpose + bf16 convert: out[c*R + r] = bf16(in[r*C + c])
// ---------------------------------------------------------------------------
__global__ void transpose_to_bf16(const float* __restrict__ in,
                                  unsigned short* __restrict__ out,
                                  int R, int C) {
  int idx = blockIdx.x * blockDim.x + threadIdx.x;
  if (idx >= R * C) return;
  int r = idx / C, c = idx - r * C;
  out[(size_t)c * R + r] = f2bf(in[idx]);
}

// ---------------------------------------------------------------------------
// LayerNorm (D=192): one wave per row, 6 elems/lane, shuffle reductions.
// ---------------------------------------------------------------------------
__global__ void layernorm_bf16(const float* __restrict__ x,
                               const float* __restrict__ gamma,
                               const float* __restrict__ beta,
                               unsigned short* __restrict__ out, int rows) {
  const int row = blockIdx.x * (blockDim.x >> 5) + (threadIdx.x >> 5);
  if (row >= rows) return;
  const int lane = threadIdx.x & 31;
  const float* xr = x + (size_t)row * 192;
  float v[6];
  float s = 0.f;
#pragma unroll
  for (int i = 0; i < 6; ++i) { v[i] = xr[lane + 32 * i]; s += v[i]; }
#pragma unroll
  for (int o = 16; o > 0; o >>= 1) s += __shfl_xor(s, o, 32);
  const float mu = s * (1.0f / 192.0f);
  float vs = 0.f;
#pragma unroll
  for (int i = 0; i < 6; ++i) { float d = v[i] - mu; vs += d * d; }
#pragma unroll
  for (int o = 16; o > 0; o >>= 1) vs += __shfl_xor(vs, o, 32);
  const float r = rsqrtf(vs * (1.0f / 192.0f) + 1e-3f);
  unsigned short* orow = out + (size_t)row * 192;
#pragma unroll
  for (int i = 0; i < 6; ++i) {
    const int c = lane + 32 * i;
    orow[c] = f2bf((v[i] - mu) * r * gamma[c] + beta[c]);
  }
}

// ---------------------------------------------------------------------------
// Generic WMMA GEMM: C[M,N] = act(A[M,K] @ B[K,N] + bias) + resid
// A bf16 row-major, BT = B^T bf16 row-major [N,K]. Wave computes 16x64 tile.
// ---------------------------------------------------------------------------
template <bool OUT_BF16, bool TRANS_OUT, bool HAS_BIAS, bool DO_GELU, bool HAS_RESID>
__global__ __launch_bounds__(128, 1) void gemm_wmma(
    const unsigned short* __restrict__ A,
    const unsigned short* __restrict__ BT,
    const float* __restrict__ bias,
    const float* __restrict__ resid,
    unsigned short* __restrict__ outB,
    float* __restrict__ outF,
    int M, int N, int K) {
  const int wid = blockIdx.x * (blockDim.x >> 5) + (threadIdx.x >> 5);
  const int nGroups = N >> 6;
  const int rowTile = wid / nGroups, cg = wid % nGroups;
  const int row0 = rowTile << 4;
  if (row0 >= M) return;
  const int lane = threadIdx.x & 31;
  const int n = lane & 15, hf = lane >> 4;
  v8f acc[4];
  const v8f vz = VZERO8;
#pragma unroll
  for (int t = 0; t < 4; ++t) acc[t] = vz;

  for (int kk = 0; kk < K; kk += 32) {
    v16bf a = load_a_frag(A, K, row0, kk);
#pragma unroll
    for (int t = 0; t < 4; ++t) {
      v16bf b = load_b_frag(BT, K, (cg << 6) + (t << 4), kk);
      acc[t] = wmma_bf16(a, b, acc[t]);
    }
  }

#pragma unroll
  for (int t = 0; t < 4; ++t) {
    const int col = (cg << 6) + (t << 4) + n;
    const float bs = HAS_BIAS ? bias[col] : 0.0f;
    Pack8 pk;
#pragma unroll
    for (int i = 0; i < 8; ++i) {
      const int row = row0 + (hf << 3) + i;   // C elem i -> row i + 8*half
      float v = acc[t][i] + bs;
      if constexpr (DO_GELU) v = 0.5f * v * (1.0f + erff(v * 0.7071067811865475f));
      if constexpr (HAS_RESID) v += resid[(size_t)row * N + col];
      if constexpr (OUT_BF16) {
        if constexpr (TRANS_OUT) pk.h[i] = f2bf(v);
        else                     outB[(size_t)row * N + col] = f2bf(v);
      } else {
        outF[(size_t)row * N + col] = v;
      }
    }
    if constexpr (OUT_BF16 && TRANS_OUT) {
      *(v4u*)(outB + (size_t)col * M + row0 + (hf << 3)) = pk.q;
    }
  }
}

// ---------------------------------------------------------------------------
// Cooperative 16x576 bf16 tile load into LDS (block of 128 threads).
// ---------------------------------------------------------------------------
__device__ __forceinline__ void load_qtile(unsigned short (*qTile)[QLD],
                                           const unsigned short* qb, int t0) {
#pragma unroll
  for (int i = 0; i < 9; ++i) {
    const int c = threadIdx.x + i * 128;      // 1152 chunks of 8 elems
    const int row = c / 72, col = (c % 72) * 8;
    *(v4u*)&qTile[row][col] = *(const v4u*)(qb + (size_t)(t0 + row) * 576 + col);
  }
}

// ---------------------------------------------------------------------------
// Attention pass 1: per-row softmax stats (m, l) of pre-mixed scores.
// Block = 4 waves sharing one (b, 16-row t-tile); wave w streams s in
// [w*512, w*512+512); partial (m,l) merged through LDS.
// ---------------------------------------------------------------------------
__global__ __launch_bounds__(128, 1) void attn_softmax_stats(
    const unsigned short* __restrict__ q,
    const unsigned short* __restrict__ k,
    const float* __restrict__ pre_w,
    float* __restrict__ mOut,
    float* __restrict__ lOut) {
  const int blk = blockIdx.x;                  // 0..1023
  const int b = blk >> 7;
  const int t0 = (blk & 127) << 4;
  const int w = threadIdx.x >> 5;
  const int lane = threadIdx.x & 31;
  const int hf = lane >> 4;
  __shared__ __align__(16) unsigned short qTile[16][QLD];
  __shared__ float mPart[4][3][16], lPart[4][3][16];

  const unsigned short* qb = q + (size_t)b * 2048 * 576;
  const unsigned short* kb = k + (size_t)b * 2048 * 576;
  load_qtile(qTile, qb, t0);
  __syncthreads();

  float pw[9];
#pragma unroll
  for (int i = 0; i < 9; ++i) pw[i] = pre_w[i];
  const float scale = 0.07216878364870323f;  // 1/sqrt(192)
  float mSt[3][8], lSt[3][8];
#pragma unroll
  for (int g = 0; g < 3; ++g)
#pragma unroll
    for (int i = 0; i < 8; ++i) { mSt[g][i] = -3.0e38f; lSt[g][i] = 0.f; }
  const v8f vz = VZERO8;

  const int sBeg = w << 9, sEnd = sBeg + 512;
  for (int s0 = sBeg; s0 < sEnd; s0 += 16) {
    NO_LICM();   // keep per-iteration ds_load of Q fragments (no hoist/spill)
    if (s0 + 16 < sEnd)  // prefetch next K s-tile row for this lane
      __builtin_prefetch(kb + (size_t)(s0 + 16 + (lane & 15)) * 576, 0, 1);
    v8f sc[3];
#pragma unroll
    for (int h = 0; h < 3; ++h) {
      v8f a = vz;
#pragma unroll
      for (int kk = 0; kk < 192; kk += 32) {
        v16bf af = load_a_frag_sh(&qTile[0][0], QLD, h * 192 + kk);
        v16bf bf = load_b_frag(kb, 576, s0, h * 192 + kk);
        a = wmma_bf16(af, bf, a);
      }
      sc[h] = a;
    }
#pragma unroll
    for (int g = 0; g < 3; ++g) {
#pragma unroll
      for (int i = 0; i < 8; ++i) {
        float sm = (sc[0][i] * pw[g] + sc[1][i] * pw[3 + g] + sc[2][i] * pw[6 + g]) * scale;
        float mx = sm;   // row max across the 16 lanes holding this row
        mx = fmaxf(mx, __shfl_xor(mx, 1, 32));
        mx = fmaxf(mx, __shfl_xor(mx, 2, 32));
        mx = fmaxf(mx, __shfl_xor(mx, 4, 32));
        mx = fmaxf(mx, __shfl_xor(mx, 8, 32));
        const float mnew = fmaxf(mSt[g][i], mx);
        float p = __expf(sm - mnew);
        p += __shfl_xor(p, 1, 32);
        p += __shfl_xor(p, 2, 32);
        p += __shfl_xor(p, 4, 32);
        p += __shfl_xor(p, 8, 32);
        lSt[g][i] = lSt[g][i] * __expf(mSt[g][i] - mnew) + p;
        mSt[g][i] = mnew;
      }
    }
  }
  if ((lane & 15) == 0) {
#pragma unroll
    for (int g = 0; g < 3; ++g)
#pragma unroll
      for (int i = 0; i < 8; ++i) {
        mPart[w][g][(hf << 3) + i] = mSt[g][i];
        lPart[w][g][(hf << 3) + i] = lSt[g][i];
      }
  }
  __syncthreads();
  const int tid = threadIdx.x;
  if (tid < 48) {
    const int g = tid >> 4, tt = tid & 15;
    float m = -3.0e38f;
#pragma unroll
    for (int ww = 0; ww < 4; ++ww) m = fmaxf(m, mPart[ww][g][tt]);
    float l = 0.f;
#pragma unroll
    for (int ww = 0; ww < 4; ++ww) l += lPart[ww][g][tt] * __expf(mPart[ww][g][tt] - m);
    mOut[((size_t)b * 3 + g) * 2048 + t0 + tt] = m;
    lOut[((size_t)b * 3 + g) * 2048 + t0 + tt] = l;
  }
}

// ---------------------------------------------------------------------------
// Attention pass 2: recompute scores, pre-mix, normalize with (m,l),
// post-mix, route pmix tiles through LDS (C-layout -> A-layout), PV matmul.
// Block = 4 waves = one (b, 16-row t-tile). Waves 0/1 build the two 16-col
// pmix sub-tiles; all 4 waves own 48 KD output columns each.
// ---------------------------------------------------------------------------
__global__ __launch_bounds__(128, 1) void attn_pv(
    const unsigned short* __restrict__ q,
    const unsigned short* __restrict__ k,
    const unsigned short* __restrict__ vT,  // [576][16384]
    const float* __restrict__ pre_w,
    const float* __restrict__ post_w,
    const float* __restrict__ mIn,
    const float* __restrict__ lIn,
    unsigned short* __restrict__ ctx) {
  const int blk = blockIdx.x;          // 0..1023
  const int b = blk >> 7;
  const int t0 = (blk & 127) << 4;
  const int w = threadIdx.x >> 5;
  const int lane = threadIdx.x & 31;
  const int n = lane & 15, hf = lane >> 4;
  __shared__ __align__(16) unsigned short qTile[16][QLD];
  __shared__ __align__(16) unsigned short pm[3][16][PMLD];

  const unsigned short* qb = q + (size_t)b * 2048 * 576;
  const unsigned short* kb = k + (size_t)b * 2048 * 576;
  load_qtile(qTile, qb, t0);

  const float scale = 0.07216878364870323f;
  float pw[9], qw[9];
#pragma unroll
  for (int i = 0; i < 9; ++i) { pw[i] = pre_w[i]; qw[i] = post_w[i]; }

  float mSt[3][8], il[3][8];
  if (w < 2) {
#pragma unroll
    for (int g = 0; g < 3; ++g)
#pragma unroll
      for (int i = 0; i < 8; ++i) {
        const size_t idx = ((size_t)b * 3 + g) * 2048 + t0 + (hf << 3) + i;
        mSt[g][i] = mIn[idx];
        const float lv = lIn[idx];
        il[g][i] = (lv > 0.f) ? 1.0f / lv : 0.f;
      }
  }

  v8f acc[3][3];
  const v8f vz = VZERO8;
#pragma unroll
  for (int j = 0; j < 3; ++j)
#pragma unroll
    for (int nt = 0; nt < 3; ++nt) acc[j][nt] = vz;
  __syncthreads();  // qTile ready

  for (int s0 = 0; s0 < 2048; s0 += 32) {
    NO_LICM();
    if (w < 2) {
      const int ss = s0 + (w << 4);
      if (s0 + 32 < 2048)
        __builtin_prefetch(kb + (size_t)(ss + 32 + (lane & 15)) * 576, 0, 1);
      v8f sc[3];
#pragma unroll
      for (int h = 0; h < 3; ++h) {
        v8f a = vz;
#pragma unroll
        for (int kk = 0; kk < 192; kk += 32) {
          v16bf af = load_a_frag_sh(&qTile[0][0], QLD, h * 192 + kk);
          v16bf bf = load_b_frag(kb, 576, ss, h * 192 + kk);
          a = wmma_bf16(af, bf, a);
        }
        sc[h] = a;
      }
      float pg[3][8];
#pragma unroll
      for (int g = 0; g < 3; ++g)
#pragma unroll
        for (int i = 0; i < 8; ++i) {
          float sm = (sc[0][i] * pw[g] + sc[1][i] * pw[3 + g] + sc[2][i] * pw[6 + g]) * scale;
          pg[g][i] = __expf(sm - mSt[g][i]) * il[g][i];
        }
#pragma unroll
      for (int j = 0; j < 3; ++j)
#pragma unroll
        for (int i = 0; i < 8; ++i) {
          const float pv = pg[0][i] * qw[j] + pg[1][i] * qw[3 + j] + pg[2][i] * qw[6 + j];
          pm[j][(hf << 3) + i][(w << 4) + n] = f2bf(pv);
        }
    }
    __syncthreads();
#pragma unroll
    for (int j = 0; j < 3; ++j) {
      v16bf a = load_a_frag_sh(&pm[j][0][0], PMLD, 0);
#pragma unroll
      for (int nt = 0; nt < 3; ++nt) {
        v16bf bf = load_b_frag(vT, 16384, j * 192 + w * 48 + nt * 16, b * 2048 + s0);
        acc[j][nt] = wmma_bf16(a, bf, acc[j][nt]);
      }
    }
    __syncthreads();
  }

#pragma unroll
  for (int j = 0; j < 3; ++j)
#pragma unroll
    for (int nt = 0; nt < 3; ++nt)
#pragma unroll
      for (int i = 0; i < 8; ++i) {
        const size_t row = (size_t)b * 2048 + t0 + (hf << 3) + i;
        const int col = j * 192 + w * 48 + nt * 16 + n;
        ctx[row * 576 + col] = f2bf(acc[j][nt][i]);
      }
}

// ---------------------------------------------------------------------------
// Host-side launch orchestration
// ---------------------------------------------------------------------------
extern "C" void kernel_launch(void* const* d_in, const int* in_sizes, int n_in,
                              void* d_out, int out_size, void* d_ws, size_t ws_size,
                              hipStream_t stream) {
  (void)in_sizes; (void)n_in; (void)out_size; (void)ws_size;
  const float* x      = (const float*)d_in[0];
  const float* gamma1 = (const float*)d_in[1];
  const float* beta1  = (const float*)d_in[2];
  const float* gamma2 = (const float*)d_in[3];
  const float* beta2  = (const float*)d_in[4];
  const float* wq     = (const float*)d_in[5];
  const float* wk     = (const float*)d_in[6];
  const float* wv     = (const float*)d_in[7];
  const float* wo     = (const float*)d_in[8];
  const float* pre_w  = (const float*)d_in[9];
  const float* post_w = (const float*)d_in[10];
  const float* w1     = (const float*)d_in[11];
  const float* b1     = (const float*)d_in[12];
  const float* w2     = (const float*)d_in[13];
  const float* b2     = (const float*)d_in[14];
  float* out = (float*)d_out;

  char* ws = (char*)d_ws;
  size_t off = 0;
  auto alloc = [&](size_t bytes) -> char* {
    char* p = ws + off;
    off = (off + bytes + 255) & ~(size_t)255;
    return p;
  };
  const int BT = 16384, D = 192, HK = 576, HID = 768;
  unsigned short* wqT = (unsigned short*)alloc((size_t)HK * D * 2);
  unsigned short* wkT = (unsigned short*)alloc((size_t)HK * D * 2);
  unsigned short* wvT = (unsigned short*)alloc((size_t)HK * D * 2);
  unsigned short* woT = (unsigned short*)alloc((size_t)D * HK * 2);
  unsigned short* w1T = (unsigned short*)alloc((size_t)HID * D * 2);
  unsigned short* w2T = (unsigned short*)alloc((size_t)D * HID * 2);
  unsigned short* nB  = (unsigned short*)alloc((size_t)BT * D * 2);
  unsigned short* qB  = (unsigned short*)alloc((size_t)BT * HK * 2);
  unsigned short* kB  = (unsigned short*)alloc((size_t)BT * HK * 2);
  unsigned short* vTB = (unsigned short*)alloc((size_t)HK * BT * 2);
  float*          mB  = (float*)alloc((size_t)8 * 3 * 2048 * 4);
  float*          lB  = (float*)alloc((size_t)8 * 3 * 2048 * 4);
  unsigned short* ctx = (unsigned short*)alloc((size_t)BT * HK * 2);
  float*          x1  = (float*)alloc((size_t)BT * D * 4);
  unsigned short* n2B = (unsigned short*)alloc((size_t)BT * D * 2);
  unsigned short* hB  = (unsigned short*)alloc((size_t)BT * HID * 2);

  // 0) weight transposes -> bf16
  transpose_to_bf16<<<(D * HK + 255) / 256, 256, 0, stream>>>(wq, wqT, D, HK);
  transpose_to_bf16<<<(D * HK + 255) / 256, 256, 0, stream>>>(wk, wkT, D, HK);
  transpose_to_bf16<<<(D * HK + 255) / 256, 256, 0, stream>>>(wv, wvT, D, HK);
  transpose_to_bf16<<<(HK * D + 255) / 256, 256, 0, stream>>>(wo, woT, HK, D);
  transpose_to_bf16<<<(D * HID + 255) / 256, 256, 0, stream>>>(w1, w1T, D, HID);
  transpose_to_bf16<<<(HID * D + 255) / 256, 256, 0, stream>>>(w2, w2T, HID, D);

  // 1) LN1
  layernorm_bf16<<<BT / 8, 256, 0, stream>>>(x, gamma1, beta1, nB, BT);

  // 2) QKV projections (V written transposed [HK][BT] for PV B-fragments)
  gemm_wmma<true,  false, false, false, false><<<2304, 128, 0, stream>>>(
      nB, wqT, nullptr, nullptr, qB, nullptr, BT, HK, D);
  gemm_wmma<true,  false, false, false, false><<<2304, 128, 0, stream>>>(
      nB, wkT, nullptr, nullptr, kB, nullptr, BT, HK, D);
  gemm_wmma<true,  true,  false, false, false><<<2304, 128, 0, stream>>>(
      nB, wvT, nullptr, nullptr, vTB, nullptr, BT, HK, D);

  // 3) attention: pass 1 (stats), pass 2 (probs + PV)
  attn_softmax_stats<<<1024, 128, 0, stream>>>(qB, kB, pre_w, mB, lB);
  attn_pv<<<1024, 128, 0, stream>>>(qB, kB, vTB, pre_w, post_w, mB, lB, ctx);

  // 4) output projection + residual -> x1 (f32)
  gemm_wmma<false, false, false, false, true><<<768, 128, 0, stream>>>(
      ctx, woT, nullptr, x, nullptr, x1, BT, D, HK);

  // 5) LN2
  layernorm_bf16<<<BT / 8, 256, 0, stream>>>(x1, gamma2, beta2, n2B, BT);

  // 6) MLP: fc1 + exact GELU -> h (bf16); fc2 + bias + residual -> out (f32)
  gemm_wmma<true,  false, true,  true,  false><<<3072, 128, 0, stream>>>(
      n2B, w1T, b1, nullptr, hB, nullptr, BT, HID, D);
  gemm_wmma<false, false, true,  false, true><<<768, 128, 0, stream>>>(
      hB, w2T, b2, x1, nullptr, out, BT, D, HID);
}
